// EmbeddingBlock_LG_78151224918203
// MI455X (gfx1250) — matrix operand order repaired and checked by reading, not compiled, not run on hardware
//
#include <hip/hip_runtime.h>
#include <hip/hip_bf16.h>
#include <math.h>

typedef __attribute__((ext_vector_type(16))) __bf16 v16bf;
typedef __attribute__((ext_vector_type(8)))  __bf16 v8bf;
typedef __attribute__((ext_vector_type(4)))  __bf16 v4bf;
typedef __attribute__((ext_vector_type(2)))  __bf16 v2bf;
typedef __attribute__((ext_vector_type(8)))  float  v8f;

#define EMB   128
#define K3    384           // 3*EMB
#define APAD  392           // padded LDS row stride (bf16 elems): 196 DWORDs -> conflict-free
#define NR    6
#define TILE_M 128          // edges per block (8 waves x 16 edges)

// ---------------------------------------------------------------------------
// Main kernel: m = silu( [feat[src] | feat[dst] | silu(rbf@Wrbf^T+brbf)] @ Wd^T + bd )
// GEMM M=E, N=128, K=384 via v_wmma_f32_16x16x32_bf16 (fp32 accumulate).
// Two N-tiles in flight -> independent WMMA chains, no hazard NOPs.
// ---------------------------------------------------------------------------
__global__ __launch_bounds__(256)
void edge_mlp_wmma_kernel(const int* __restrict__ Z, const int* __restrict__ src,
                          const int* __restrict__ dst, const float* __restrict__ rbf,
                          const float* __restrict__ emb, const float* __restrict__ Wrbf,
                          const float* __restrict__ brbf, const float* __restrict__ Wd,
                          const float* __restrict__ bd, float* __restrict__ out, int E)
{
    extern __shared__ char smem[];
    __bf16* sA  = (__bf16*)smem;                               // [128][392] bf16  (98 KB)
    __bf16* sW  = (__bf16*)(smem + TILE_M * APAD * 2);         // [128][392] bf16  (98 KB)
    float*  sWr = (float*)(smem + 2 * TILE_M * APAD * 2);      // [128][6]
    float*  sBr = sWr + EMB * NR;                              // [128]
    float*  sBd = sBr + EMB;                                   // [128]

    const int tid  = threadIdx.x;
    const int wave = tid >> 5;
    const int lane = tid & 31;

    // --- stage small tables -------------------------------------------------
    for (int i = tid; i < EMB * NR; i += 256) sWr[i] = Wrbf[i];
    for (int i = tid; i < EMB; i += 256) { sBr[i] = brbf[i]; sBd[i] = bd[i]; }
    __syncthreads();

    // --- stage W_dense as bf16 (float4 reads -> one ds_store_b64 each) -------
    const float4* Wd4 = (const float4*)Wd;
    for (int i4 = tid; i4 < EMB * (K3 / 4); i4 += 256) {
        float4 w = Wd4[i4];
        int j = i4 / (K3 / 4);
        int k = (i4 % (K3 / 4)) * 4;
        v4bf p;
        p[0] = (__bf16)w.x; p[1] = (__bf16)w.y; p[2] = (__bf16)w.z; p[3] = (__bf16)w.w;
        *(v4bf*)(sW + j * APAD + k) = p;
    }

    // --- stage A tile: each wave builds its own 16 edge rows -----------------
    // 2 adjacent columns per lane per trip: float2 loads, packed b32 DS stores.
    const long eb = (long)blockIdx.x * TILE_M + wave * 16;
    for (int le = 0; le < 16; ++le) {
        long e = eb + le;
        __bf16* arow = sA + (wave * 16 + le) * APAD;
        if (e < (long)E) {
            int zs = Z[src[e]], zt = Z[dst[e]];
            const float* fs = emb + (long)zs * EMB;
            const float* ft = emb + (long)zt * EMB;
            float r0 = rbf[e * NR + 0], r1 = rbf[e * NR + 1], r2 = rbf[e * NR + 2];
            float r3 = rbf[e * NR + 3], r4 = rbf[e * NR + 4], r5 = rbf[e * NR + 5];
            for (int c = lane * 2; c < K3; c += 64) {   // 6 trips, branch-uniform
                float v0, v1;
                if (c < EMB)          { v0 = fs[c];        v1 = fs[c + 1]; }
                else if (c < 2 * EMB) { v0 = ft[c - EMB];  v1 = ft[c - EMB + 1]; }
                else {
                    int j = c - 2 * EMB;
                    const float* w0 = sWr + j * NR;
                    const float* w1 = w0 + NR;
                    float a0 = sBr[j]     + r0 * w0[0] + r1 * w0[1] + r2 * w0[2]
                                          + r3 * w0[3] + r4 * w0[4] + r5 * w0[5];
                    float a1 = sBr[j + 1] + r0 * w1[0] + r1 * w1[1] + r2 * w1[2]
                                          + r3 * w1[3] + r4 * w1[4] + r5 * w1[5];
                    v0 = a0 / (1.0f + __expf(-a0));      // silu
                    v1 = a1 / (1.0f + __expf(-a1));
                }
                v2bf p; p[0] = (__bf16)v0; p[1] = (__bf16)v1;
                *(v2bf*)(arow + c) = p;
            }
        } else {
            for (int c = lane * 2; c < K3; c += 64) {
                v2bf p; p[0] = (__bf16)0.0f; p[1] = (__bf16)0.0f;
                *(v2bf*)(arow + c) = p;
            }
        }
    }
    __syncthreads();

    // --- load A fragments per ISA 16-bit A layout, keep resident -------------
    const int r    = lane & 15;
    const int hi16 = lane >> 4;                       // lanes 16-31 take K-base +8
    const __bf16* pa = sA + (wave * 16 + r) * APAD + (hi16 ? 8 : 0);
    v16bf Af[12];
#pragma unroll
    for (int ks = 0; ks < 12; ++ks) {
        v8bf lo = *(const v8bf*)(pa + 32 * ks);       // K = k0+{0..7}/{8..15}
        v8bf hi = *(const v8bf*)(pa + 32 * ks + 16);  // K = k0+16+{0..7}/{8..15}
        Af[ks] = __builtin_shufflevector(lo, hi, 0,1,2,3,4,5,6,7,8,9,10,11,12,13,14,15);
    }

    // --- 4 x (2 N-tiles) x 12 K-steps of WMMA, dual independent accumulators -
    for (int nt = 0; nt < 8; nt += 2) {
        v8f C0 = {};
        v8f C1 = {};
        const __bf16* pb0 = sW + (nt * 16 + r) * APAD + (hi16 ? 8 : 0);  // B K-base 0/8
        const __bf16* pb1 = pb0 + 16 * APAD;
#pragma unroll
        for (int ks = 0; ks < 12; ++ks) {
            v8bf lo0 = *(const v8bf*)(pb0 + 32 * ks);
            v8bf hi0 = *(const v8bf*)(pb0 + 32 * ks + 16);
            v8bf lo1 = *(const v8bf*)(pb1 + 32 * ks);
            v8bf hi1 = *(const v8bf*)(pb1 + 32 * ks + 16);
            v16bf B0 = __builtin_shufflevector(lo0, hi0, 0,1,2,3,4,5,6,7,8,9,10,11,12,13,14,15);
            v16bf B1 = __builtin_shufflevector(lo1, hi1, 0,1,2,3,4,5,6,7,8,9,10,11,12,13,14,15);
            C0 = __builtin_amdgcn_wmma_f32_16x16x32_bf16(
                     false, Af[ks], false, B0, (short)0, C0, false, false);
            C1 = __builtin_amdgcn_wmma_f32_16x16x32_bf16(
                     false, Af[ks], false, B1, (short)0, C1, false, false);
        }
        // epilogue: C layout — VGPR i: M = i + (lane<16 ? 0 : 8), N = lane%16
        int   col0 = nt * 16 + r;
        int   col1 = col0 + 16;
        float b0 = sBd[col0];
        float b1 = sBd[col1];
        long  ebase = eb + (hi16 ? 8 : 0);
#pragma unroll
        for (int i = 0; i < 8; ++i) {
            long erow = ebase + i;
            if (erow < (long)E) {
                float v = C0[i] + b0;
                v = v / (1.0f + __expf(-v));          // silu
                __builtin_nontemporal_store(v, &out[erow * EMB + col0]);
                float u = C1[i] + b1;
                u = u / (1.0f + __expf(-u));
                __builtin_nontemporal_store(u, &out[erow * EMB + col1]);
            }
        }
    }
}

// ---------------------------------------------------------------------------
// rbf_env = envelope(x) * sin(n*pi*x)/x, n=1..6, p=6 (a=-28, b=48, c=-21)
// ---------------------------------------------------------------------------
__global__ __launch_bounds__(256)
void rbf_env_kernel(const float* __restrict__ d, float* __restrict__ out, int E)
{
    int e = blockIdx.x * blockDim.x + threadIdx.x;
    if (e >= E) return;
    const float PI = 3.14159265358979323846f;
    float x   = d[e] * 0.2f;                 // d / CUTOFF
    float inv = 1.0f / x;
    float x2 = x * x, x4 = x2 * x2, x5 = x4 * x, x6 = x5 * x, x7 = x6 * x;
    float env = inv - 28.0f * x5 + 48.0f * x6 - 21.0f * x7;
    float s   = env * inv;                   // envelope(x)/x
    float t = PI * x, st, ct;
    sincosf(t, &st, &ct);
    float twoc = 2.0f * ct;
    float sn_1 = 0.0f, sn = st;              // sin(n*t) recurrence
    float* o = out + (long)e * NR;
#pragma unroll
    for (int n = 1; n <= NR; ++n) {
        __builtin_nontemporal_store(s * sn, &o[n - 1]);
        float nx = twoc * sn - sn_1;
        sn_1 = sn; sn = nx;
    }
}

// ---------------------------------------------------------------------------
extern "C" void kernel_launch(void* const* d_in, const int* in_sizes, int n_in,
                              void* d_out, int out_size, void* d_ws, size_t ws_size,
                              hipStream_t stream)
{
    const int*   Z    = (const int*)d_in[0];
    const int*   src  = (const int*)d_in[1];
    const int*   dst  = (const int*)d_in[2];
    const float* rbf  = (const float*)d_in[3];
    const float* dvec = (const float*)d_in[4];
    const float* emb  = (const float*)d_in[5];
    const float* Wrbf = (const float*)d_in[6];
    const float* brbf = (const float*)d_in[7];
    const float* Wd   = (const float*)d_in[8];
    const float* bd   = (const float*)d_in[9];
    int E = in_sizes[1];                       // src count = number of edges
    float* out = (float*)d_out;

    const size_t shmem = (size_t)(2 * TILE_M * APAD * 2)        // sA + sW (bf16)
                       + (size_t)(EMB * NR + EMB + EMB) * 4;    // sWr + sBr + sBd
    (void)hipFuncSetAttribute((const void*)edge_mlp_wmma_kernel,
                              hipFuncAttributeMaxDynamicSharedMemorySize, (int)shmem);

    int blocks = (E + TILE_M - 1) / TILE_M;
    edge_mlp_wmma_kernel<<<blocks, 256, shmem, stream>>>(
        Z, src, dst, rbf, emb, Wrbf, brbf, Wd, bd, out, E);

    rbf_env_kernel<<<(E + 255) / 256, 256, 0, stream>>>(
        dvec, out + (size_t)E * EMB, E);
}
